// PairwiseScore_42451456753705
// MI455X (gfx1250) — compile-verified, matrix operand order checked by reading, not run on hardware
//
#include <hip/hip_runtime.h>
#include <hip/hip_bf16.h>

// ---------------------------------------------------------------------------
// PairwiseScore on MI455X (gfx1250, wave32, WMMA bf16 w/ f32 accumulate)
//
//   out[i,j] = (ms[i] + ms[j] + MLP([g_i, g_j, g_i*g_j])) / 3
//
// Decomposition:  pair@W1 = g_i@W1a + g_j@W1b + (g_i*g_j)@W1c
//   -> A_i = g@W1a + b1  and  B_j = g@W1b precomputed once (512x160 f32)
//   -> per-pair work: two bf16 WMMA GEMMs, K=256 then K=160 (150 padded)
// LDS is laid out manually with H1s at offset 0 so every H1 ds access folds
// its p*320 row offset into the 16-bit DS immediate (no per-row address VALU).
// ---------------------------------------------------------------------------

#define NSPAN 512
#define EDIM  256
#define HID   150
#define HPAD  160

typedef __attribute__((ext_vector_type(4)))  float          v4f;
typedef __attribute__((ext_vector_type(8)))  float          v8f;
typedef __attribute__((ext_vector_type(4)))  unsigned short v4u16;
typedef __attribute__((ext_vector_type(8)))  __bf16         v8bf;
typedef __attribute__((ext_vector_type(16))) __bf16         v16bf;

__device__ __forceinline__ unsigned short f2bf(float x) {
  return __builtin_bit_cast(unsigned short, (__bf16)x);
}

__device__ __forceinline__ v16bf join16(v8bf lo, v8bf hi) {
  v16bf r;
#pragma unroll
  for (int t = 0; t < 8; ++t) { r[t] = lo[t]; r[t + 8] = hi[t]; }
  return r;
}

// Workspace layout (bytes)
#define WS_W1CT 0u            // bf16 [HPAD][EDIM]  (transposed W1c, zero-padded)
#define WS_W2T  81920u        // bf16 [HPAD][HPAD]  (transposed W2, zero-padded)
#define WS_W3P  133120u       // f32  [HPAD]
#define WS_B2P  133760u       // f32  [HPAD]
#define WS_AROW 134400u       // f32  [NSPAN][HPAD]  g@W1a + b1
#define WS_BROW 462080u       // f32  [NSPAN][HPAD]  g@W1b

// LDS arena layout (bytes) -- H1s first so its DS offsets fit in immediates
#define L_H1    0u                         // bf16 [128][HPAD]   40960 B
#define L_APAIR 40960u                     // bf16 [128][EDIM]   65536 B
#define L_AROW  (40960u + 65536u)          // f32  [8][HPAD]      5120 B
#define L_BROW  (L_AROW + 5120u)           // f32  [16][HPAD]    10240 B
#define L_W3    (L_BROW + 10240u)          // f32  [HPAD]          640 B
#define L_B2    (L_W3 + 640u)              // f32  [HPAD]          640 B
#define L_TOTAL (L_B2 + 640u)              // = 123136 B

// ---------------------------------------------------------------------------
// Kernel 0: weight transpose/convert + per-span row tables (tiny FLOP count)
// ---------------------------------------------------------------------------
__global__ void prep_kernel(const float* __restrict__ emb,
                            const float* __restrict__ W1,
                            const float* __restrict__ b1,
                            const float* __restrict__ W2,
                            const float* __restrict__ b2,
                            const float* __restrict__ W3,
                            unsigned short* __restrict__ w1ct,
                            unsigned short* __restrict__ w2t,
                            float* __restrict__ w3p,
                            float* __restrict__ b2p,
                            float* __restrict__ arow,
                            float* __restrict__ brow) {
  const int tid = blockIdx.x * blockDim.x + threadIdx.x;
  const int nth = gridDim.x * blockDim.x;

  // W1c^T : [n][k], n = hidden col, k = E index of the product block
  for (int idx = tid; idx < HPAD * EDIM; idx += nth) {
    const int n = idx >> 8, k = idx & 255;
    const float v = (n < HID) ? W1[(2 * EDIM + k) * HID + n] : 0.0f;
    w1ct[idx] = f2bf(v);
  }
  // W2^T : [n][k] padded to 160x160 with zeros
  for (int idx = tid; idx < HPAD * HPAD; idx += nth) {
    const int n = idx / HPAD, k = idx % HPAD;
    const float v = (n < HID && k < HID) ? W2[k * HID + n] : 0.0f;
    w2t[idx] = f2bf(v);
  }
  for (int h = tid; h < HPAD; h += nth) {
    w3p[h] = (h < HID) ? W3[h] : 0.0f;
    b2p[h] = (h < HID) ? b2[h] : 0.0f;
  }
  // A_i = g@W1a + b1 ; B_j = g@W1b   (f32, exact)
  for (int idx = tid; idx < NSPAN * HPAD; idx += nth) {
    const int i = idx / HPAD, h = idx % HPAD;
    float sa = 0.0f, sb = 0.0f;
    if (h < HID) {
      sa = b1[h];
      const float* gr = emb + i * EDIM;
#pragma unroll 4
      for (int e = 0; e < EDIM; ++e) {
        const float g = gr[e];
        sa = fmaf(g, W1[e * HID + h], sa);
        sb = fmaf(g, W1[(EDIM + e) * HID + h], sb);
      }
    }
    arow[idx] = sa;
    brow[idx] = sb;
  }
}

// ---------------------------------------------------------------------------
// Kernel 1: per-pair MLP. Block = 8 i x 16 j = 128 pairs, 256 threads (8 waves)
// Each wave owns one 16-row M-tile across all 10 N-tiles.
// ---------------------------------------------------------------------------
__global__ void __launch_bounds__(256)
pair_mlp_kernel(const float* __restrict__ emb,
                const float* __restrict__ ms,
                const float* __restrict__ b3,
                const unsigned short* __restrict__ w1ct,
                const unsigned short* __restrict__ w2t,
                const float* __restrict__ w3p,
                const float* __restrict__ b2p,
                const float* __restrict__ arow,
                const float* __restrict__ brow,
                float* __restrict__ out) {
  __shared__ __align__(16) unsigned char smem[L_TOTAL];
  unsigned short* const H1s   = (unsigned short*)(smem + L_H1);
  unsigned short* const Apair = (unsigned short*)(smem + L_APAIR);
  float* const arowS = (float*)(smem + L_AROW);
  float* const browS = (float*)(smem + L_BROW);
  float* const w3S   = (float*)(smem + L_W3);
  float* const b2S   = (float*)(smem + L_B2);

  const int tid  = threadIdx.x;
  const int lane = tid & 31;
  const int half = lane >> 4;   // wave32 lane half selects K sub-chunk
  const int ln15 = lane & 15;
  const int mt   = tid >> 5;    // wave id == M-tile id (0..7)
  const int j0   = blockIdx.x * 16;
  const int i0   = blockIdx.y * 8;

  __builtin_prefetch(w1ct, 0, 1); // global_prefetch_b8: warm weight lines

  // ---- Phase A: stage pair products (bf16) + row tables into LDS ----------
  const v4f* embv = (const v4f*)emb;
  for (int idx = tid; idx < 128 * (EDIM / 4); idx += 256) {
    const int p  = idx >> 6;       // pair 0..127  (p = ii*16 + jj)
    const int e4 = idx & 63;       // group of 4 elements
    const int gi = i0 + (p >> 4);
    const int gj = j0 + (p & 15);
    const v4f a = embv[gi * (EDIM / 4) + e4];
    const v4f b = embv[gj * (EDIM / 4) + e4];
    v4u16 r;
#pragma unroll
    for (int t = 0; t < 4; ++t) r[t] = f2bf(a[t] * b[t]);
    *(v4u16*)&Apair[p * EDIM + e4 * 4] = r;
  }
  for (int idx = tid; idx < 8 * HPAD; idx += 256)
    arowS[idx] = arow[i0 * HPAD + idx];            // rows i0..i0+7 contiguous
  for (int idx = tid; idx < 16 * HPAD; idx += 256)
    browS[idx] = brow[j0 * HPAD + idx];            // rows j0..j0+15 contiguous
  if (tid < HPAD) { w3S[tid] = w3p[tid]; b2S[tid] = b2p[tid]; }
  __syncthreads();

  // C-fragment row ids, invariant across N-tiles: VGPR v -> pair row v+8*half
  int iiOff[8], jjOff[8];
#pragma unroll
  for (int v = 0; v < 8; ++v) {
    const int p = mt * 16 + v + half * 8;
    iiOff[v] = (p >> 4) * HPAD;   // into arowS
    jjOff[v] = (p & 15) * HPAD;   // into browS
  }

  // ---- Phase B: GEMM1  H1 = relu( Apair @ W1c^T + A_i + B_j ) -------------
  const int afr = mt * 16 + ln15;             // A-fragment source row
#pragma unroll
  for (int nt = 0; nt < 10; ++nt) {
    v8f acc = {0.f, 0.f, 0.f, 0.f, 0.f, 0.f, 0.f, 0.f};
    const int n = nt * 16 + ln15;
#pragma unroll
    for (int ks = 0; ks < 8; ++ks) {
      const int kb = ks * 32;
      const v8bf alo = *(const v8bf*)&Apair[afr * EDIM + kb + half * 8];
      const v8bf ahi = *(const v8bf*)&Apair[afr * EDIM + kb + 16 + half * 8];
      const v8bf blo = *(const v8bf*)&w1ct[n * EDIM + kb + half * 8];
      const v8bf bhi = *(const v8bf*)&w1ct[n * EDIM + kb + 16 + half * 8];
      acc = __builtin_amdgcn_wmma_f32_16x16x32_bf16(
          false, join16(alo, ahi), false, join16(blo, bhi),
          (short)0, acc, false, false);
    }
    // epilogue: pure LDS traffic; H1 offsets fold into DS immediates
#pragma unroll
    for (int v = 0; v < 8; ++v) {
      const int p = mt * 16 + v + half * 8;
      float x = acc[v] + arowS[iiOff[v] + n] + browS[jjOff[v] + n];
      x = fmaxf(x, 0.0f);
      H1s[p * HPAD + n] = f2bf(x);
    }
  }
  __syncthreads();

  // ---- Phase C: GEMM2 + fused W3 dot --------------------------------------
  float sdot[8];
#pragma unroll
  for (int v = 0; v < 8; ++v) sdot[v] = 0.0f;

#pragma unroll
  for (int nt = 0; nt < 10; ++nt) {
    v8f acc = {0.f, 0.f, 0.f, 0.f, 0.f, 0.f, 0.f, 0.f};
    const int n = nt * 16 + ln15;
#pragma unroll
    for (int ks = 0; ks < 5; ++ks) {
      const int kb = ks * 32;
      const v8bf alo = *(const v8bf*)&H1s[afr * HPAD + kb + half * 8];
      const v8bf ahi = *(const v8bf*)&H1s[afr * HPAD + kb + 16 + half * 8];
      const v8bf blo = *(const v8bf*)&w2t[n * HPAD + kb + half * 8];
      const v8bf bhi = *(const v8bf*)&w2t[n * HPAD + kb + 16 + half * 8];
      acc = __builtin_amdgcn_wmma_f32_16x16x32_bf16(
          false, join16(alo, ahi), false, join16(blo, bhi),
          (short)0, acc, false, false);
    }
    const float w3n = w3S[n];
    const float b2n = b2S[n];
#pragma unroll
    for (int v = 0; v < 8; ++v) {
      const float h2 = fmaxf(acc[v] + b2n, 0.0f);
      sdot[v] = fmaf(h2, w3n, sdot[v]);
    }
  }

  // reduce the 16 n-lanes of each wave32 half (xor 1,2,4,8 stays in-half)
#pragma unroll
  for (int v = 0; v < 8; ++v) {
    float s = sdot[v];
    s += __shfl_xor(s, 1);
    s += __shfl_xor(s, 2);
    s += __shfl_xor(s, 4);
    s += __shfl_xor(s, 8);
    sdot[v] = s;
  }

  const float b3v = b3[0];
  if (ln15 == 0) {
#pragma unroll
    for (int v = 0; v < 8; ++v) {
      const int p  = mt * 16 + v + half * 8;
      const int gi = i0 + (p >> 4);
      const int gj = j0 + (p & 15);
      const float s = sdot[v] + b3v;
      out[gi * NSPAN + gj] = (ms[gi] + ms[gj] + s) * (1.0f / 3.0f);
    }
  }
}

// ---------------------------------------------------------------------------
extern "C" void kernel_launch(void* const* d_in, const int* in_sizes, int n_in,
                              void* d_out, int out_size, void* d_ws, size_t ws_size,
                              hipStream_t stream) {
  const float* emb = (const float*)d_in[0]; // (1,512,256)
  const float* ms  = (const float*)d_in[1]; // (1,512,1)
  const float* W1  = (const float*)d_in[2]; // (768,150)
  const float* b1  = (const float*)d_in[3]; // (150,)
  const float* W2  = (const float*)d_in[4]; // (150,150)
  const float* b2  = (const float*)d_in[5]; // (150,)
  const float* W3  = (const float*)d_in[6]; // (150,1)
  const float* b3  = (const float*)d_in[7]; // (1,)
  float* out = (float*)d_out;               // (1,512,512,1)

  char* ws = (char*)d_ws;
  unsigned short* w1ct = (unsigned short*)(ws + WS_W1CT);
  unsigned short* w2t  = (unsigned short*)(ws + WS_W2T);
  float* w3p  = (float*)(ws + WS_W3P);
  float* b2p  = (float*)(ws + WS_B2P);
  float* arow = (float*)(ws + WS_AROW);
  float* brow = (float*)(ws + WS_BROW);

  prep_kernel<<<dim3(64), dim3(256), 0, stream>>>(
      emb, W1, b1, W2, b2, W3, w1ct, w2t, w3p, b2p, arow, brow);

  pair_mlp_kernel<<<dim3(NSPAN / 16, NSPAN / 8), dim3(256), 0, stream>>>(
      emb, ms, b3, w1ct, w2t, w3p, b2p, arow, brow, out);
}
